// SelfAttention_12369505812808
// MI455X (gfx1250) — matrix operand order repaired and checked by reading, not compiled, not run on hardware
//
#include <hip/hip_runtime.h>
#include <hip/hip_bf16.h>

#define EMBED    1024
#define HEADS    16
#define HEAD_DIM 64
#define NBATCH   2
#define SEQ      2048
#define M_TOTAL  (NBATCH * SEQ)   // 4096 flattened rows

typedef __attribute__((ext_vector_type(16))) __bf16 v16bf;
typedef __attribute__((ext_vector_type(8)))  __bf16 v8bf;
typedef __attribute__((ext_vector_type(8)))  float  v8f;

__device__ __forceinline__ v8f wmma_bf16(v16bf a, v16bf b, v8f c) {
    // D = A(16x32 bf16) * B(32x16 bf16) + C(16x16 f32)
    return __builtin_amdgcn_wmma_f32_16x16x32_bf16(
        false, a, false, b, (short)0, c, false, false);
}

// ---- bf16 fragment loaders (layouts per CDNA5 ISA 7.12.2) ----
// Bases are wave-uniform; lane-variant part is a 32-bit element offset so the
// compiler can use GVS addressing (SGPR base + VGPR 32-bit offset), one VGPR/stream.

// A fragment 16x32: lane L: row = m0 + L%16 ; k runs {(L/16)*8..+7} and {16+(L/16)*8..+7}
__device__ __forceinline__ v16bf load_a_bf16(const __bf16* __restrict__ A, int lda, int m0, int k0) {
    int lane = threadIdx.x & 31;
    int off = (m0 + (lane & 15)) * lda + k0 + ((lane >> 4) << 3);
    v8bf lo = *(const v8bf*)(A + off);
    v8bf hi = *(const v8bf*)(A + off + 16);
    return __builtin_shufflevector(lo, hi, 0,1,2,3,4,5,6,7,8,9,10,11,12,13,14,15);
}

// B fragment 32x16 with B[k][n] = W[n0+n][k0+k] (W row-major): lane L: n = L%16,
// k runs {(L/16)*8..+7} and {16+(L/16)*8..+7} -> two 16B loads per lane.
__device__ __forceinline__ v16bf load_bT_bf16(const __bf16* __restrict__ W, int ldw, int n0, int k0) {
    int lane = threadIdx.x & 31;
    int off = (n0 + (lane & 15)) * ldw + k0 + ((lane >> 4) << 3);
    v8bf lo = *(const v8bf*)(W + off);
    v8bf hi = *(const v8bf*)(W + off + 16);
    return __builtin_shufflevector(lo, hi, 0,1,2,3,4,5,6,7,8,9,10,11,12,13,14,15);
}

// -------------------- kernel 0: f32 -> bf16 pre-convert --------------------
__global__ void cvt_f32_bf16_kernel(const float* __restrict__ src, __bf16* __restrict__ dst, int n4) {
    int i = (blockIdx.x * blockDim.x + threadIdx.x);
    if (i < n4) {
        float4 f = ((const float4*)src)[i];
        __bf16* d = dst + (size_t)i * 4;
        d[0] = (__bf16)f.x; d[1] = (__bf16)f.y; d[2] = (__bf16)f.z; d[3] = (__bf16)f.w;
    }
}

// -------------------- kernel 1: fused Q/K/V projection --------------------
// grid (M_TOTAL/32, EMBED/64, 3), block 32: one wave -> 32x64 tile (2x4 = 8 WMMA/k-step,
// 64 acc VGPRs -> spill-free). Q,K -> [N][H][S][D]; V -> transposed [N][H][D][S].
__global__ void qkv_proj_kernel(const __bf16* __restrict__ xb,
                                const __bf16* __restrict__ wqkv,
                                __bf16* __restrict__ qw,
                                __bf16* __restrict__ kw,
                                __bf16* __restrict__ vtw) {
    int m0 = blockIdx.x * 32, n0 = blockIdx.y * 64, which = blockIdx.z;
    const __bf16* W = wqkv + (size_t)which * EMBED * EMBED;
    v8f acc[2][4] = {};
    for (int k0 = 0; k0 < EMBED; k0 += 32) {
        v16bf a[2], b[4];
#pragma unroll
        for (int mi = 0; mi < 2; ++mi) a[mi] = load_a_bf16(xb, EMBED, m0 + mi * 16, k0);
#pragma unroll
        for (int nj = 0; nj < 4; ++nj) b[nj] = load_bT_bf16(W, EMBED, n0 + nj * 16, k0);
        if (k0 + 32 < EMBED) {
            __builtin_prefetch(xb + m0 * EMBED + k0 + 32, 0, 1);
            __builtin_prefetch(W + n0 * EMBED + k0 + 32, 0, 1);
        }
#pragma unroll
        for (int mi = 0; mi < 2; ++mi)
#pragma unroll
            for (int nj = 0; nj < 4; ++nj)
                acc[mi][nj] = wmma_bf16(a[mi], b[nj], acc[mi][nj]);
    }
    int lane = threadIdx.x & 31;
    // 32-row tile never crosses batch boundary (2048 % 32 == 0)
    int nb = m0 >> 11;
    int s_base = (m0 & (SEQ - 1)) + 8 * (lane >> 4);
    if (which == 2) {
        // V transposed: [N][H][D][S]
        __bf16* vbase = vtw + ((size_t)nb * HEADS) * HEAD_DIM * SEQ;
#pragma unroll
        for (int nj = 0; nj < 4; ++nj) {
            int o = n0 + nj * 16 + (lane & 15);
            int roff = o * SEQ + s_base;   // (h*64+d)*SEQ + s, o==h*64+d
#pragma unroll
            for (int mi = 0; mi < 2; ++mi)
#pragma unroll
                for (int i = 0; i < 8; ++i)
                    vbase[roff + mi * 16 + i] = (__bf16)acc[mi][nj][i];
        }
    } else {
        __bf16* dst = (which == 0) ? qw : kw;   // [N][H][S][D]
        __bf16* dbase = dst + ((size_t)nb * HEADS) * SEQ * HEAD_DIM;
#pragma unroll
        for (int nj = 0; nj < 4; ++nj) {
            int o = n0 + nj * 16 + (lane & 15);
            int h = o >> 6, d = o & 63;
            int hoff = h * (SEQ * HEAD_DIM) + d + s_base * HEAD_DIM;
#pragma unroll
            for (int mi = 0; mi < 2; ++mi)
#pragma unroll
                for (int i = 0; i < 8; ++i)
                    dbase[hoff + (mi * 16 + i) * HEAD_DIM] = (__bf16)acc[mi][nj][i];
        }
    }
}

// -------------------- kernel 2: flash attention --------------------
// grid (SEQ/16, HEADS, NBATCH), block 32 (one wave owns a 16-query tile).
__global__ void attn_kernel(const __bf16* __restrict__ qw,
                            const __bf16* __restrict__ kw,
                            const __bf16* __restrict__ vtw,
                            __bf16* __restrict__ attn_out /* [N,S,E] bf16 */) {
    __shared__ float ptile[16 * 36];   // stride 36: rows 16B-aligned, conflict-free
    int qt = blockIdx.x, h = blockIdx.y, n = blockIdx.z;
    const __bf16* Qp = qw  + ((size_t)n * HEADS + h) * SEQ * HEAD_DIM;  // [s][d]
    const __bf16* Kp = kw  + ((size_t)n * HEADS + h) * SEQ * HEAD_DIM;  // [s][d]
    const __bf16* Vt = vtw + ((size_t)n * HEADS + h) * HEAD_DIM * SEQ;  // [d][s]
    int lane = threadIdx.x & 31;

    v16bf qa0 = load_a_bf16(Qp, HEAD_DIM, qt * 16, 0);
    v16bf qa1 = load_a_bf16(Qp, HEAD_DIM, qt * 16, 32);

    float mrow[8], lrow[8];
#pragma unroll
    for (int i = 0; i < 8; ++i) { mrow[i] = -1e30f; lrow[i] = 0.0f; }
    v8f oacc[4] = {};
    const float scale = 0.03125f;   // 1/sqrt(1024)

    for (int kc = 0; kc < SEQ; kc += 32) {
        if (kc + 32 < SEQ) {
            __builtin_prefetch(Kp + (kc + 32) * HEAD_DIM, 0, 1);
            __builtin_prefetch(Vt + kc + 32, 0, 1);
        }
        // scores: two 16x16 tiles; B[k=d][n=key] = K[key][d]
        v8f s0 = {}, s1 = {};
        {
            v16bf b0 = load_bT_bf16(Kp, HEAD_DIM, kc,      0);
            v16bf b1 = load_bT_bf16(Kp, HEAD_DIM, kc,      32);
            v16bf b2 = load_bT_bf16(Kp, HEAD_DIM, kc + 16, 0);
            v16bf b3 = load_bT_bf16(Kp, HEAD_DIM, kc + 16, 32);
            s0 = wmma_bf16(qa0, b0, s0);
            s0 = wmma_bf16(qa1, b1, s0);
            s1 = wmma_bf16(qa0, b2, s1);
            s1 = wmma_bf16(qa1, b3, s1);
        }
        // online softmax (row r = i + 8*(lane/16) spans the 16-lane half-group)
        float alpha[8];
#pragma unroll
        for (int i = 0; i < 8; ++i) {
            float a0 = s0[i] * scale;
            float a1 = s1[i] * scale;
            float v = fmaxf(a0, a1);
            for (int off = 1; off < 16; off <<= 1)
                v = fmaxf(v, __shfl_xor(v, off, 32));
            float nm = fmaxf(mrow[i], v);
            alpha[i] = __expf(mrow[i] - nm);
            mrow[i] = nm;
            float p0 = __expf(a0 - nm);
            float p1 = __expf(a1 - nm);
            s0[i] = p0; s1[i] = p1;
            float ps = p0 + p1;
            for (int off = 1; off < 16; off <<= 1)
                ps += __shfl_xor(ps, off, 32);
            lrow[i] = lrow[i] * alpha[i] + ps;
        }
        // stage P (C-layout f32) -> LDS -> reread as A fragment (bf16)
        int mloc = 8 * (lane >> 4);
#pragma unroll
        for (int i = 0; i < 8; ++i) {
            ptile[(mloc + i) * 36 + (lane & 15)]      = s0[i];
            ptile[(mloc + i) * 36 + 16 + (lane & 15)] = s1[i];
        }
        __syncthreads();
        v16bf pa;
        {
            const float* pr = &ptile[(lane & 15) * 36];
            int kh = (lane >> 4) * 8;
            float4 f0 = *(const float4*)(pr + kh);
            float4 f1 = *(const float4*)(pr + kh + 4);
            float4 f2 = *(const float4*)(pr + 16 + kh);
            float4 f3 = *(const float4*)(pr + 16 + kh + 4);
            pa[0]  = (__bf16)f0.x; pa[1]  = (__bf16)f0.y; pa[2]  = (__bf16)f0.z; pa[3]  = (__bf16)f0.w;
            pa[4]  = (__bf16)f1.x; pa[5]  = (__bf16)f1.y; pa[6]  = (__bf16)f1.z; pa[7]  = (__bf16)f1.w;
            pa[8]  = (__bf16)f2.x; pa[9]  = (__bf16)f2.y; pa[10] = (__bf16)f2.z; pa[11] = (__bf16)f2.w;
            pa[12] = (__bf16)f3.x; pa[13] = (__bf16)f3.y; pa[14] = (__bf16)f3.z; pa[15] = (__bf16)f3.w;
        }
        // O = O*alpha + P(16x32) * V(32x64); B[k=key][n=d] = Vt[d][key] (contiguous in key)
#pragma unroll
        for (int dt = 0; dt < 4; ++dt) {
#pragma unroll
            for (int i = 0; i < 8; ++i) oacc[dt][i] *= alpha[i];
            v16bf vb = load_bT_bf16(Vt, SEQ, dt * 16, kc);
            oacc[dt] = wmma_bf16(pa, vb, oacc[dt]);
        }
        __syncthreads();
    }
    // normalize and store bf16 [N,S,E]
    __bf16* obase = attn_out + ((size_t)n * SEQ) * EMBED + h * HEAD_DIM;
#pragma unroll
    for (int dt = 0; dt < 4; ++dt) {
#pragma unroll
        for (int i = 0; i < 8; ++i) {
            int s = qt * 16 + i + 8 * (lane >> 4);
            int d = dt * 16 + (lane & 15);
            obase[s * EMBED + d] = (__bf16)(oacc[dt][i] / lrow[i]);
        }
    }
}

// -------------------- kernel 3: output projection + bias --------------------
// grid (M_TOTAL/32, EMBED/64), block 32 (32x64 tile per wave, spill-free).
__global__ void out_proj_kernel(const __bf16* __restrict__ attn,
                                const __bf16* __restrict__ wob,
                                const float* __restrict__ bo,
                                float* __restrict__ out) {
    int m0 = blockIdx.x * 32, n0 = blockIdx.y * 64;
    v8f acc[2][4] = {};
    for (int k0 = 0; k0 < EMBED; k0 += 32) {
        v16bf a[2], b[4];
#pragma unroll
        for (int mi = 0; mi < 2; ++mi) a[mi] = load_a_bf16(attn, EMBED, m0 + mi * 16, k0);
#pragma unroll
        for (int nj = 0; nj < 4; ++nj) b[nj] = load_bT_bf16(wob, EMBED, n0 + nj * 16, k0);
        if (k0 + 32 < EMBED) {
            __builtin_prefetch(attn + m0 * EMBED + k0 + 32, 0, 1);
            __builtin_prefetch(wob + n0 * EMBED + k0 + 32, 0, 1);
        }
#pragma unroll
        for (int mi = 0; mi < 2; ++mi)
#pragma unroll
            for (int nj = 0; nj < 4; ++nj)
                acc[mi][nj] = wmma_bf16(a[mi], b[nj], acc[mi][nj]);
    }
    int lane = threadIdx.x & 31;
    int m_base = m0 + 8 * (lane >> 4);
#pragma unroll
    for (int nj = 0; nj < 4; ++nj) {
        int o = n0 + nj * 16 + (lane & 15);
        float bias = bo[o];
        int ooff = m_base * EMBED + o;
#pragma unroll
        for (int mi = 0; mi < 2; ++mi)
#pragma unroll
            for (int i = 0; i < 8; ++i)
                out[ooff + (mi * 16 + i) * EMBED] = acc[mi][nj][i] + bias;
    }
}

extern "C" void kernel_launch(void* const* d_in, const int* in_sizes, int n_in,
                              void* d_out, int out_size, void* d_ws, size_t ws_size,
                              hipStream_t stream) {
    (void)in_sizes; (void)n_in; (void)out_size; (void)ws_size;
    const float* x  = (const float*)d_in[0];
    const float* Wq = (const float*)d_in[1];
    const float* Wk = (const float*)d_in[2];
    const float* Wv = (const float*)d_in[3];
    const float* Wo = (const float*)d_in[4];
    const float* bo = (const float*)d_in[5];
    float* out = (float*)d_out;

    const size_t EL_X = (size_t)M_TOTAL * EMBED;   // 4,194,304
    const size_t EL_W = (size_t)EMBED * EMBED;     // 1,048,576

    __bf16* xb   = (__bf16*)d_ws;
    __bf16* wqkv = xb + EL_X;          // 3 weight matrices, bf16
    __bf16* wob  = wqkv + 3 * EL_W;
    __bf16* qw   = wob + EL_W;         // [N,H,S,D]
    __bf16* kw   = qw + EL_X;          // [N,H,S,D]
    __bf16* vtw  = kw + EL_X;          // [N,H,D,S] (transposed V)
    __bf16* attn = vtw + EL_X;         // [N,S,E]

    // one-shot f32 -> bf16 pre-conversion (removes cvt + waits from hot loops)
    cvt_f32_bf16_kernel<<<(int)(EL_X / 4 / 256), 256, 0, stream>>>(x,  xb,              (int)(EL_X / 4));
    cvt_f32_bf16_kernel<<<(int)(EL_W / 4 / 256), 256, 0, stream>>>(Wq, wqkv,            (int)(EL_W / 4));
    cvt_f32_bf16_kernel<<<(int)(EL_W / 4 / 256), 256, 0, stream>>>(Wk, wqkv + EL_W,     (int)(EL_W / 4));
    cvt_f32_bf16_kernel<<<(int)(EL_W / 4 / 256), 256, 0, stream>>>(Wv, wqkv + 2 * EL_W, (int)(EL_W / 4));
    cvt_f32_bf16_kernel<<<(int)(EL_W / 4 / 256), 256, 0, stream>>>(Wo, wob,             (int)(EL_W / 4));

    dim3 g1(M_TOTAL / 32, EMBED / 64, 3);
    qkv_proj_kernel<<<g1, 32, 0, stream>>>(xb, wqkv, qw, kw, vtw);

    dim3 g2(SEQ / 16, HEADS, NBATCH);
    attn_kernel<<<g2, 32, 0, stream>>>(qw, kw, vtw, attn);

    dim3 g3(M_TOTAL / 32, EMBED / 64);
    out_proj_kernel<<<g3, 32, 0, stream>>>(attn, wob, bo, out);
}
